// MoeHashBlock_21973052686543
// MI455X (gfx1250) — matrix operand order repaired and checked by reading, not compiled, not run on hardware
//
#include <hip/hip_runtime.h>
#include <hip/hip_fp16.h>

typedef __attribute__((ext_vector_type(16))) _Float16 v16h;
typedef __attribute__((ext_vector_type(8)))  float    v8f;
typedef __attribute__((ext_vector_type(4)))  int      v4i;

#define WMMA_F16(a,b,c) __builtin_amdgcn_wmma_f32_16x16x32_f16(false,(a),false,(b),(short)0,(c),false,false)

constexpr int Bsz = 4, S = 2048, D = 1024, E = 8, F = 4096, H = 16, HD = 64;
constexpr int R = Bsz * S;   // 8192 tokens

// ---------------- CDNA5 async global->LDS copy (ASYNCcnt-tracked), with fallback
#if defined(__has_builtin)
#if __has_builtin(__builtin_amdgcn_global_load_async_to_lds_b128)
#define HAVE_ASYNC_LDS 1
#endif
#endif

#ifdef HAVE_ASYNC_LDS
typedef __attribute__((address_space(1))) v4i* gv4_ptr;   // HIP prints AS1 as __device__
typedef __attribute__((address_space(3))) v4i* lv4_ptr;   // HIP prints AS3 as __shared__
__device__ inline void async_b128(const void* g, void* l) {
  __builtin_amdgcn_global_load_async_to_lds_b128((gv4_ptr)(void*)g, (lv4_ptr)l, 0, 0);
}
__device__ inline void async_wait0() {
#if __has_builtin(__builtin_amdgcn_s_wait_asynccnt)
  __builtin_amdgcn_s_wait_asynccnt(0);
#else
  asm volatile("s_wait_asynccnt 0" ::: "memory");
#endif
}
#endif

// ---------------------------------------------------------------- utilities

__global__ void cvt_f32_f16(const float* __restrict__ in, _Float16* __restrict__ out, int n) {
  int i = blockIdx.x * blockDim.x + threadIdx.x;
  int stride = gridDim.x * blockDim.x;
  for (; i < n; i += stride) out[i] = (_Float16)in[i];
}

__global__ __launch_bounds__(256) void rmsnorm_kernel(const float* __restrict__ x,
                                                      const float* __restrict__ w,
                                                      _Float16* __restrict__ out) {
  __shared__ float red[256];
  int row = blockIdx.x;
  const float* xr = x + (size_t)row * D;
  float ss = 0.f;
  for (int i = threadIdx.x; i < D; i += 256) { float v = xr[i]; ss += v * v; }
  red[threadIdx.x] = ss;
  __syncthreads();
  for (int o = 128; o > 0; o >>= 1) {
    if (threadIdx.x < o) red[threadIdx.x] += red[threadIdx.x + o];
    __syncthreads();
  }
  float rstd = rsqrtf(red[0] * (1.0f / D) + 1e-6f);
  _Float16* orow = out + (size_t)row * D;
  for (int i = threadIdx.x; i < D; i += 256) orow[i] = (_Float16)(xr[i] * rstd * w[i]);
}

__device__ inline float gelu_tanh(float x) {
  return 0.5f * x * (1.0f + tanhf(0.7978845608f * (x + 0.044715f * x * x * x)));
}

// CDNA5 16-bit A-matrix 16x32 fragment from LDS (row-major tile):
// lane = {row = lane&15, kq = lane>>4}; element i -> k = kq*16 + i
__device__ inline v16h lds_afrag(const _Float16* base, int stride, int lane) {
  int row = lane & 15, kq = lane >> 4;
  v16h f;
#pragma unroll
  for (int i = 0; i < 16; i++) {
    int k = kq * 16 + i;
    f[i] = base[row * stride + k];
  }
  return f;
}

// CDNA5 16-bit B-matrix 32x16 fragment from LDS (row-major [k][n] tile):
// lane = {col = lane&15, kb = (lane>>4)*16}; element i -> k = kb + i
__device__ inline v16h lds_bfrag(const _Float16* base, int stride, int lane) {
  int col = lane & 15, kb = (lane >> 4) * 16;
  v16h f;
#pragma unroll
  for (int i = 0; i < 16; i++) f[i] = base[(kb + i) * stride + col];
  return f;
}

// ---------------------------------------------------------------- generic WMMA GEMM
// Block tile 64x128, 8 waves (2x4), each wave owns 32x32 (2x2 WMMA tiles).
// MODE 0: C = A*B               -> outF            (QKV projection)
// MODE 1: C = A*B + resid       -> outF            (WO projection + residual)
// MODE 2: C = gelu(A[idx]*B[e]) -> outH (gathered) (MoE up-proj)
// MODE 3: outF[idx] += A*B[e]                      (MoE down-proj, scatter-add)
template <int MODE>
__global__ __launch_bounds__(256) void gemm_wmma(
    const _Float16* __restrict__ A, const _Float16* __restrict__ Bw,
    int M, int N, int K,
    const float* __restrict__ resid, float* __restrict__ outF, _Float16* __restrict__ outH,
    const int* __restrict__ counts, const int* __restrict__ offsets, const int* __restrict__ idx) {
  __shared__ _Float16 As[64 * 40];    // 64 rows x 32 halves (+8 pad)
  __shared__ _Float16 Bs[32 * 136];   // 32 k-rows x 128 cols (+8 pad)
  int t = threadIdx.x, lane = t & 31, wave = t >> 5;
  int wr = wave >> 2, wc = wave & 3;
  int rowTile = blockIdx.y, colTile = blockIdx.x;
  int e = blockIdx.z;
  int cnt = M;
  if (MODE >= 2) {
    cnt = counts[e];
    if (rowTile * 64 >= cnt) return;
  }
  const _Float16* Bp = Bw + (size_t)e * K * N + (size_t)colTile * 128;

  // staging coordinates: A 64x32 (8 halves/thread), B 32x128 (16 halves/thread)
  int ar = t >> 2, ac = (t & 3) * 8;
  int br = t >> 3, bc = (t & 7) * 16;

  // resolve A row pointer once (row constant across K loop)
  int slot = rowTile * 64 + ar;
  const _Float16* aptr;
  if (MODE == 2) {
    int s2 = slot < cnt ? slot : cnt - 1;
    aptr = A + (size_t)idx[(size_t)e * M + s2] * K;
  } else if (MODE == 3) {
    int s2 = slot < cnt ? slot : cnt - 1;
    aptr = A + (size_t)(offsets[e] + s2) * K;
  } else {
    aptr = A + (size_t)slot * K;
  }

  v8f acc[2][2] = {{{}, {}}, {{}, {}}};
  for (int k0 = 0; k0 < K; k0 += 32) {
    __syncthreads();
#ifdef HAVE_ASYNC_LDS
    async_b128(aptr + k0 + ac, &As[ar * 40 + ac]);
    async_b128(Bp + (size_t)(k0 + br) * N + bc, &Bs[br * 136 + bc]);
    async_b128(Bp + (size_t)(k0 + br) * N + bc + 8, &Bs[br * 136 + bc + 8]);
    async_wait0();
#else
    *(uint4*)&As[ar * 40 + ac] = *(const uint4*)(aptr + k0 + ac);
    *(uint4*)&Bs[br * 136 + bc] = *(const uint4*)(Bp + (size_t)(k0 + br) * N + bc);
    *(uint4*)&Bs[br * 136 + bc + 8] = *(const uint4*)(Bp + (size_t)(k0 + br) * N + bc + 8);
#endif
    __syncthreads();
    v16h a0 = lds_afrag(As + (wr * 32) * 40, 40, lane);
    v16h a1 = lds_afrag(As + (wr * 32 + 16) * 40, 40, lane);
    v16h b0 = lds_bfrag(Bs + wc * 32, 136, lane);
    v16h b1 = lds_bfrag(Bs + wc * 32 + 16, 136, lane);
    acc[0][0] = WMMA_F16(a0, b0, acc[0][0]);
    acc[0][1] = WMMA_F16(a0, b1, acc[0][1]);
    acc[1][0] = WMMA_F16(a1, b0, acc[1][0]);
    acc[1][1] = WMMA_F16(a1, b1, acc[1][1]);
  }

  // C layout: element r -> row = (lane>>4)*8 + r, col = lane&15
  int lr = lane >> 4, lc = lane & 15;
#pragma unroll
  for (int ai = 0; ai < 2; ai++)
#pragma unroll
    for (int bi = 0; bi < 2; bi++)
#pragma unroll
      for (int r = 0; r < 8; r++) {
        int rrow = rowTile * 64 + wr * 32 + ai * 16 + lr * 8 + r;
        int col = colTile * 128 + wc * 32 + bi * 16 + lc;
        float v = acc[ai][bi][r];
        if (MODE == 0) {
          outF[(size_t)rrow * N + col] = v;
        } else if (MODE == 1) {
          outF[(size_t)rrow * N + col] = v + resid[(size_t)rrow * N + col];
        } else if (MODE == 2) {
          if (rrow < cnt) outH[(size_t)(offsets[e] + rrow) * N + col] = (_Float16)gelu_tanh(v);
        } else {
          if (rrow < cnt) {
            int tok = idx[(size_t)e * M + rrow];
            outF[(size_t)tok * N + col] += v;
          }
        }
      }
}

// ---------------------------------------------------------------- RoPE + head repack
__global__ __launch_bounds__(256) void rope_repack(const float* __restrict__ qkv,
                                                   _Float16* __restrict__ q16,
                                                   _Float16* __restrict__ k16,
                                                   _Float16* __restrict__ v16) {
  int gid = blockIdx.x * 256 + threadIdx.x;  // R*H*32 threads
  if (gid >= R * H * 32) return;
  int j = gid & 31;
  int h = (gid >> 5) & 15;
  int tok = gid >> 9;
  int b = tok / S, s = tok % S;
  const float* base = qkv + (size_t)tok * 3 * D;
  float q1 = base[h * HD + j],         q2 = base[h * HD + 32 + j];
  float k1 = base[D + h * HD + j],     k2 = base[D + h * HD + 32 + j];
  float v1 = base[2 * D + h * HD + j], v2 = base[2 * D + h * HD + 32 + j];
  // inv_freq = 10000^(-j/32) = exp(-ln(1e4)/32 * j)
  float inv = __expf(-0.2878231366f * (float)j);
  float sn, c;
  __sincosf((float)s * inv, &sn, &c);
  size_t o = ((size_t)(b * H + h) * S + s) * HD + j;
  q16[o] = (_Float16)(q1 * c - q2 * sn);  q16[o + 32] = (_Float16)(q2 * c + q1 * sn);
  k16[o] = (_Float16)(k1 * c - k2 * sn);  k16[o + 32] = (_Float16)(k2 * c + k1 * sn);
  v16[o] = (_Float16)v1;                  v16[o + 32] = (_Float16)v2;
}

// ---------------------------------------------------------------- flash attention
// grid: (S/128, B*H); block 256 = 8 waves; each wave owns 16 query rows x hd=64.
__global__ __launch_bounds__(256) void flash_attn(const _Float16* __restrict__ q16,
                                                  const _Float16* __restrict__ k16,
                                                  const _Float16* __restrict__ v16,
                                                  _Float16* __restrict__ attnOut) {
  __shared__ _Float16 Kt[32 * 72];
  __shared__ _Float16 Vt[32 * 72];
  __shared__ _Float16 Pl[8 * 16 * 40];
  int t = threadIdx.x, lane = t & 31, wave = t >> 5;
  int bh = blockIdx.y, b = bh >> 4, h = bh & 15;
  int qb = blockIdx.x;
  int qw = qb * 128 + wave * 16;  // wave's first query (s index)
  int lr = lane >> 4, lc = lane & 15;
  const _Float16* qbase = q16 + ((size_t)bh * S + qw) * HD;
  const _Float16* kbase = k16 + (size_t)bh * S * HD;
  const _Float16* vbase = v16 + (size_t)bh * S * HD;

  // Q fragments: 2 x (16x32) covering hd=64, straight from global (A layout)
  v16h qf[2];
#pragma unroll
  for (int f = 0; f < 2; f++)
#pragma unroll
    for (int i = 0; i < 16; i++) {
      int d = f * 32 + lr * 16 + i;
      qf[f][i] = qbase[(size_t)lc * HD + d];
    }

  v8f acc[4] = {{}, {}, {}, {}};
  float m[8], l[8];
#pragma unroll
  for (int r = 0; r < 8; r++) { m[r] = -3.0e38f; l[r] = 0.f; }

  int sr = t >> 3, sc = (t & 7) * 8;
  _Float16* Pw = Pl + wave * 16 * 40;
  int nkt = qb * 4 + 4;  // key chunks of 32 up to the block's causal frontier
  for (int kt = 0; kt < nkt; kt++) {
    __syncthreads();
#ifdef HAVE_ASYNC_LDS
    async_b128(kbase + (size_t)(kt * 32 + sr) * HD + sc, &Kt[sr * 72 + sc]);
    async_b128(vbase + (size_t)(kt * 32 + sr) * HD + sc, &Vt[sr * 72 + sc]);
    async_wait0();
#else
    *(uint4*)&Kt[sr * 72 + sc] = *(const uint4*)(kbase + (size_t)(kt * 32 + sr) * HD + sc);
    *(uint4*)&Vt[sr * 72 + sc] = *(const uint4*)(vbase + (size_t)(kt * 32 + sr) * HD + sc);
#endif
    __syncthreads();
    if (kt * 32 > qw + 15) continue;  // fully masked for this wave; barrier counts stay uniform

    // scores: 16q x 32k as two 16x16 tiles, K(=hd) split in two 32-chunks
    float sreg[2][8];
#pragma unroll
    for (int ct = 0; ct < 2; ct++) {
      v16h bf0, bf1;
#pragma unroll
      for (int i = 0; i < 16; i++) {
        int krow = (ct * 16 + lc) * 72;
        int kk = lr * 16 + i;
        bf0[i] = Kt[krow + kk];
        bf1[i] = Kt[krow + 32 + kk];
      }
      v8f sacc = {};
      sacc = WMMA_F16(qf[0], bf0, sacc);
      sacc = WMMA_F16(qf[1], bf1, sacc);
#pragma unroll
      for (int r = 0; r < 8; r++) {
        int qrow = qw + lr * 8 + r;
        int key = kt * 32 + ct * 16 + lc;
        float sv = sacc[r] * 0.125f;  // 1/sqrt(64)
        sreg[ct][r] = (key <= qrow) ? sv : -3.0e38f;
      }
    }

    // online softmax (row stats live in the 16-lane half holding that row)
#pragma unroll
    for (int r = 0; r < 8; r++) {
      float mx = fmaxf(sreg[0][r], sreg[1][r]);
      for (int o = 1; o < 16; o <<= 1) mx = fmaxf(mx, __shfl_xor(mx, o, 16));
      float mn = fmaxf(m[r], mx);
      float alpha = __expf(m[r] - mn);
      m[r] = mn;
      float p0 = __expf(sreg[0][r] - mn), p1 = __expf(sreg[1][r] - mn);
      float rs = p0 + p1;
      for (int o = 1; o < 16; o <<= 1) rs += __shfl_xor(rs, o, 16);
      l[r] = l[r] * alpha + rs;
#pragma unroll
      for (int ot = 0; ot < 4; ot++) acc[ot][r] *= alpha;
      Pw[(lr * 8 + r) * 40 + lc] = (_Float16)p0;
      Pw[(lr * 8 + r) * 40 + 16 + lc] = (_Float16)p1;
    }
    // C-layout -> A-layout transpose went through per-wave LDS; drain DS counter
    asm volatile("s_wait_dscnt 0" ::: "memory");
    v16h pf = lds_afrag(Pw, 40, lane);
#pragma unroll
    for (int ot = 0; ot < 4; ot++) {
      v16h bv;
#pragma unroll
      for (int i = 0; i < 16; i++) bv[i] = Vt[(lr * 16 + i) * 72 + ot * 16 + lc];
      acc[ot] = WMMA_F16(pf, bv, acc[ot]);
    }
  }

  // epilogue: out/l, repack heads back into [token, D] f16 for the WO GEMM
#pragma unroll
  for (int r = 0; r < 8; r++) {
    int qrow = qw + lr * 8 + r;
    float inv = 1.0f / l[r];
#pragma unroll
    for (int ot = 0; ot < 4; ot++) {
      attnOut[((size_t)b * S + qrow) * D + h * HD + ot * 16 + lc] = (_Float16)(acc[ot][r] * inv);
    }
  }
}

// ---------------------------------------------------------------- MoE routing
__global__ void route_zero(int* counts) {
  if (threadIdx.x < E) counts[threadIdx.x] = 0;
}
__global__ void route_scatter(const int* __restrict__ mt, int* counts, int* idx) {
  int i = blockIdx.x * 256 + threadIdx.x;
  if (i >= R) return;
  int e = mt[i];
  int slot = atomicAdd(&counts[e], 1);
  idx[(size_t)e * R + slot] = i;
}
__global__ void route_offsets(const int* __restrict__ counts, int* offsets) {
  if (threadIdx.x == 0) {
    int a = 0;
    for (int e = 0; e < E; e++) { offsets[e] = a; a += counts[e]; }
  }
}

// ---------------------------------------------------------------- launcher
extern "C" void kernel_launch(void* const* d_in, const int* in_sizes, int n_in,
                              void* d_out, int out_size, void* d_ws, size_t ws_size,
                              hipStream_t stream) {
  (void)in_sizes; (void)n_in; (void)out_size; (void)ws_size;
  const float* x    = (const float*)d_in[0];
  const int*   mt   = (const int*)d_in[1];
  const float* rms1 = (const float*)d_in[2];
  const float* wqkv = (const float*)d_in[3];
  const float* wo   = (const float*)d_in[4];
  const float* rms2 = (const float*)d_in[5];
  const float* w1   = (const float*)d_in[6];
  const float* w2   = (const float*)d_in[7];
  float* out = (float*)d_out;

  char* p = (char*)d_ws;
  auto alloc = [&](size_t bytes) -> char* {
    char* r = p;
    p += (bytes + 255) & ~(size_t)255;
    return r;
  };
  _Float16* wqkv16 = (_Float16*)alloc((size_t)D * 3 * D * 2);
  _Float16* wo16   = (_Float16*)alloc((size_t)D * D * 2);
  _Float16* w1h    = (_Float16*)alloc((size_t)E * D * F * 2);
  _Float16* w2h    = (_Float16*)alloc((size_t)E * F * D * 2);
  _Float16* xn16   = (_Float16*)alloc((size_t)R * D * 2);
  float*    qkv32  = (float*)alloc((size_t)R * 3 * D * 4);
  _Float16* q16    = (_Float16*)alloc((size_t)R * D * 2);
  _Float16* k16    = (_Float16*)alloc((size_t)R * D * 2);
  _Float16* v16    = (_Float16*)alloc((size_t)R * D * 2);
  float*    h32    = (float*)alloc((size_t)R * D * 4);
  int* counts  = (int*)alloc(E * 4);
  int* offsets = (int*)alloc(E * 4);
  int* idx     = (int*)alloc((size_t)E * R * 4);
  // aliases (lifetimes do not overlap)
  _Float16* attn16 = xn16;             // xn consumed by QKV GEMM before attention writes
  _Float16* y1     = (_Float16*)qkv32; // qkv consumed by rope before MoE up-proj writes
  _Float16* h2n16  = q16;              // q consumed by attention before rmsnorm2 writes

  // 1) weights fp32 -> f16
  cvt_f32_f16<<<2048, 256, 0, stream>>>(wqkv, wqkv16, D * 3 * D);
  cvt_f32_f16<<<2048, 256, 0, stream>>>(wo, wo16, D * D);
  cvt_f32_f16<<<4096, 256, 0, stream>>>(w1, w1h, E * D * F);
  cvt_f32_f16<<<4096, 256, 0, stream>>>(w2, w2h, E * F * D);

  // 2) rmsnorm1
  rmsnorm_kernel<<<R, 256, 0, stream>>>(x, rms1, xn16);

  // 3) QKV projection: [8192,1024] x [1024,3072]
  gemm_wmma<0><<<dim3(3 * D / 128, R / 64, 1), 256, 0, stream>>>(
      xn16, wqkv16, R, 3 * D, D, nullptr, qkv32, nullptr, nullptr, nullptr, nullptr);

  // 4) RoPE + head repack
  rope_repack<<<(R * H * 32) / 256, 256, 0, stream>>>(qkv32, q16, k16, v16);

  // 5) flash attention
  flash_attn<<<dim3(S / 128, Bsz * H, 1), 256, 0, stream>>>(q16, k16, v16, attn16);

  // 6) WO projection + residual: h = x + attn @ wo
  gemm_wmma<1><<<dim3(D / 128, R / 64, 1), 256, 0, stream>>>(
      attn16, wo16, R, D, D, x, h32, nullptr, nullptr, nullptr, nullptr);

  // 7) rmsnorm2
  rmsnorm_kernel<<<R, 256, 0, stream>>>(h32, rms2, h2n16);

  // 8) out = h (MoE scatter-adds on top)
  (void)hipMemcpyAsync(out, h32, (size_t)R * D * 4, hipMemcpyDeviceToDevice, stream);

  // 9) hash routing: per-expert token lists
  route_zero<<<1, 32, 0, stream>>>(counts);
  route_scatter<<<R / 256, 256, 0, stream>>>(mt, counts, idx);
  route_offsets<<<1, 32, 0, stream>>>(counts, offsets);

  // 10) MoE up-proj (gathered rows, GELU, f16)
  gemm_wmma<2><<<dim3(F / 128, R / 64, E), 256, 0, stream>>>(
      h2n16, w1h, R, F, D, nullptr, nullptr, y1, counts, offsets, idx);

  // 11) MoE down-proj (scatter-add into out)
  gemm_wmma<3><<<dim3(D / 128, R / 64, E), 256, 0, stream>>>(
      y1, w2h, R, D, F, nullptr, out, nullptr, counts, offsets, idx);
}
